// UpSampleSelfAttention_17145509446492
// MI455X (gfx1250) — matrix operand order repaired and checked by reading, not compiled, not run on hardware
//
#include <hip/hip_runtime.h>
#include <hip/hip_bf16.h>
#include <math.h>

typedef __attribute__((ext_vector_type(16))) _Float16 v16h;
typedef __attribute__((ext_vector_type(8)))  _Float16 v8h;
typedef __attribute__((ext_vector_type(8)))  float    v8f;
typedef unsigned int u32x4 __attribute__((ext_vector_type(4)));
typedef int          i32x4 __attribute__((ext_vector_type(4)));
typedef int          i32x8 __attribute__((ext_vector_type(8)));

#define BB    4
#define CIN   128
#define NSEL  1024
#define NTOT  2048
#define NH    4
#define DH    64
#define OC    256

union F16Frag { v16h v; v8h h[2]; };

static __device__ __forceinline__ v8f wmma_f16(const F16Frag& a, const F16Frag& b, v8f c) {
  // 8 args: (neg_a, A, neg_b, B, c_mod, C, reuse_a, reuse_b)
  return __builtin_amdgcn_wmma_f32_16x16x32_f16(false, a.v, false, b.v, (short)0, c, false, false);
}

// ---------------------------------------------------------------------------
// TDM: DMA a 2-D f16 tile (tile_d0 x tile_d1, row stride stride0 elements)
// from global memory into LDS at byte offset lds_off.  D# per ISA 8.3/8.4.
// ---------------------------------------------------------------------------
static __device__ __forceinline__ void tdm_load_2d(unsigned lds_off,
                                                   const void* gaddr,
                                                   unsigned tile_d0,
                                                   unsigned tile_d1,
                                                   unsigned tensor_d0,
                                                   unsigned tensor_d1,
                                                   unsigned long long stride0) {
  unsigned long long ga = (unsigned long long)(uintptr_t)gaddr;
  u32x4 g0;
  g0.x = 0x1u;                                   // count=1, user descriptor
  g0.y = lds_off;                                // lds_addr (bytes)
  g0.z = (unsigned)(ga & 0xffffffffu);           // global_addr[31:0]
  g0.w = (unsigned)((ga >> 32) & 0x01ffffffu) | (2u << 30);  // [56:32] | type=2
  i32x8 g1;
  g1[0] = (int)(1u << 16);                       // data_size=1 (2 bytes)
  g1[1] = (int)((tensor_d0 & 0xffffu) << 16);    // tensor_dim0[15:0]
  g1[2] = (int)(((tensor_d0 >> 16) & 0xffffu) | ((tensor_d1 & 0xffffu) << 16));
  g1[3] = (int)(((tensor_d1 >> 16) & 0xffffu) | ((tile_d0 & 0xffffu) << 16));
  g1[4] = (int)(tile_d1 & 0xffffu);              // tile_dim1, tile_dim2=0
  g1[5] = (int)(stride0 & 0xffffffffu);          // tensor_dim0_stride[31:0]
  g1[6] = (int)((stride0 >> 32) & 0xffffu);      // stride[47:32], stride1=0
  g1[7] = 0;
  i32x4 z4 = {};
  i32x8 z8 = {};
#if __clang_major__ >= 23
  __builtin_amdgcn_tensor_load_to_lds(g0, g1, z4, z4, z8, 0);
#else
  __builtin_amdgcn_tensor_load_to_lds(g0, g1, z4, z4, 0);
#endif
}

// ---------------------------------------------------------------------------
// Kernel 1: scatter select/drop columns into xT (b, n, c) f16, and transpose
// pcd_up into pT (b, n, c) f16.  block = 128 threads, grid = (N, B).
// ---------------------------------------------------------------------------
__global__ void usa_prep_kernel(const float* __restrict__ pcd_up,
                                const float* __restrict__ psel,
                                const float* __restrict__ pdrop,
                                const int* __restrict__ isel,
                                const int* __restrict__ idrop,
                                _Float16* __restrict__ xT,
                                _Float16* __restrict__ pT) {
  int j = blockIdx.x;
  int b = blockIdx.y;
  int c = threadIdx.x;
  float s;
  int dest;
  if (j < NSEL) {
    s = psel[((size_t)b * CIN + c) * NSEL + j];
    dest = isel[b * NSEL + j];
  } else {
    s = pdrop[((size_t)b * CIN + c) * NSEL + (j - NSEL)];
    dest = idrop[b * NSEL + (j - NSEL)];
  }
  xT[((size_t)b * NTOT + dest) * CIN + c] = (_Float16)s;
  pT[((size_t)b * NTOT + j) * CIN + c] =
      (_Float16)pcd_up[((size_t)b * CIN + c) * NTOT + j];
}

// ---------------------------------------------------------------------------
// Kernel 2: stack Wq/Wk/Wv/Wskip into one f16 matrix Wall (1024 x 128).
// ---------------------------------------------------------------------------
__global__ void usa_wcvt_kernel(const float* __restrict__ Wq,
                                const float* __restrict__ Wk,
                                const float* __restrict__ Wv,
                                const float* __restrict__ Ws,
                                _Float16* __restrict__ Wall) {
  int g = blockIdx.x * blockDim.x + threadIdx.x;   // 0 .. 1024*128-1
  int o = g >> 7;
  int c = g & 127;
  const float* src = (o < 256) ? Wq : (o < 512) ? Wk : (o < 768) ? Wv : Ws;
  Wall[g] = (_Float16)src[(size_t)(o & 255) * CIN + c];
}

// ---------------------------------------------------------------------------
// Kernel 3: fused projection GEMM.  One wave = one 16x16 output tile of
// Y = xT * Wall^T  (K=128, 4 x wmma_f32_16x16x32_f16).
// o in [0,256)   -> Q  (b,h,n,64) f16
// o in [256,512) -> K  (b,h,n,64) f16
// o in [512,768) -> V^T(b,h,64,n) f16
// o in [768,1024)-> skip (f32, written directly to d_out)
// ---------------------------------------------------------------------------
__global__ void usa_proj_kernel(const _Float16* __restrict__ xT,
                                const _Float16* __restrict__ pT,
                                const _Float16* __restrict__ Wall,
                                _Float16* __restrict__ Qw,
                                _Float16* __restrict__ Kw,
                                _Float16* __restrict__ VTw,
                                float* __restrict__ out) {
  int lane = threadIdx.x & 31;
  int tid = blockIdx.x * 8 + (threadIdx.x >> 5);  // 0 .. 32767
  int nt = tid & 127;          // point tile
  int ot = (tid >> 7) & 63;    // output-channel tile (of 1024)
  int b  = tid >> 13;
  int hi = lane >> 4;          // half-wave id
  int lm = lane & 15;

  const _Float16* src = (ot < 48) ? xT : pT;   // skip GEMM reads pcd_up
  int n0 = nt * 16 + lm;
  int o0 = ot * 16 + lm;
  const _Float16* arow = src + ((size_t)b * NTOT + n0) * CIN;
  const _Float16* brow = Wall + (size_t)o0 * CIN;

  v8f acc = {};
#pragma unroll
  for (int kc = 0; kc < 4; ++kc) {
    F16Frag a, w;
    // A (16 points x 32 ch): lanes 0-15 hold K = {0..7,16..23}, lanes 16-31 +8
    const _Float16* ap = arow + kc * 32 + hi * 8;
    a.h[0] = *(const v8h*)(ap);
    a.h[1] = *(const v8h*)(ap + 16);
    // B (32 ch x 16 out): lane holds one column, K = {0..7,16..23}, lanes 16-31 +8
    const _Float16* bp = brow + kc * 32 + hi * 8;
    w.h[0] = *(const v8h*)(bp);
    w.h[1] = *(const v8h*)(bp + 16);
    acc = wmma_f16(a, w, acc);
  }

  int o = ot * 16 + lm;        // D: lane = output-channel column
#pragma unroll
  for (int e = 0; e < 8; ++e) {
    int point = nt * 16 + e + hi * 8;   // D: vgpr e = row (+8 for upper half)
    float val = acc[e];
    if (o < 256) {
      int h = o >> 6, d = o & 63;
      Qw[(((size_t)b * NH + h) * NTOT + point) * DH + d] = (_Float16)val;
    } else if (o < 512) {
      int oo = o - 256, h = oo >> 6, d = oo & 63;
      Kw[(((size_t)b * NH + h) * NTOT + point) * DH + d] = (_Float16)val;
    } else if (o < 768) {
      int oo = o - 512, h = oo >> 6, d = oo & 63;
      VTw[(((size_t)b * NH + h) * DH + d) * NTOT + point] = (_Float16)val;
    } else {
      out[((size_t)b * OC + (o - 768)) * NTOT + point] = val;   // skip path
    }
  }
}

// ---------------------------------------------------------------------------
// Kernel 4: flash attention.  One block = one (b,h,row-group), 8 waves x 16 query rows.
// Per 32-key iteration, wave 0 DMAs the shared K tile (1-D, 4 KB) and V^T
// tile (2-D strided) into LDS via the Tensor Data Mover, waits on TENSORcnt,
// and a workgroup barrier publishes them.  Each wave: 4 WMMAs for S = Q K^T,
// online softmax via half-wave shuffles, P staged through LDS (C-layout ->
// A-layout), 4 WMMAs for O += P V.  Epilogue: out += O / l (on top of skip).
// ---------------------------------------------------------------------------
__global__ void usa_attn_kernel(const _Float16* __restrict__ Qw,
                                const _Float16* __restrict__ Kw,
                                const _Float16* __restrict__ VTw,
                                float* __restrict__ out) {
  extern __shared__ _Float16 smem[];
  _Float16* ldsK = smem;                 // 2048 halves (32 keys x 64 d)
  _Float16* ldsV = smem + 2048;          // 2048 halves (64 dims x 32 keys)
  int lane = threadIdx.x & 31;
  int w = threadIdx.x >> 5;
  _Float16* myl = smem + 4096 + w * 512; // per-wave 16x32 P staging

  int rg = blockIdx.x & 15;
  int h  = (blockIdx.x >> 4) & 3;
  int b  = blockIdx.x >> 6;
  int rb = rg * 8 + w;                   // this wave's query row-block
  int hi = lane >> 4;
  int lm = lane & 15;

  size_t bh = (size_t)b * NH + h;
  const _Float16* qbase = Qw + bh * NTOT * DH;
  const _Float16* kbase = Kw + bh * NTOT * DH;
  const _Float16* vbase = VTw + bh * DH * NTOT;

  // Q fragments for this row block: two K-chunks of 32 (head dim 64)
  F16Frag qa[2];
  {
    const _Float16* qp = qbase + (size_t)(rb * 16 + lm) * DH;
#pragma unroll
    for (int dc = 0; dc < 2; ++dc) {
      const _Float16* p = qp + dc * 32 + hi * 8;
      qa[dc].h[0] = *(const v8h*)(p);
      qa[dc].h[1] = *(const v8h*)(p + 16);
    }
  }

  v8f o_acc[4];
#pragma unroll
  for (int dt = 0; dt < 4; ++dt) o_acc[dt] = (v8f){};
  float m_run[8], l_run[8];
#pragma unroll
  for (int e = 0; e < 8; ++e) { m_run[e] = -1.0e30f; l_run[e] = 0.0f; }

  for (int kb = 0; kb < NTOT / 32; ++kb) {
    if (w == 0) {
      // K tile: contiguous 32 keys x 64 d  (1-D DMA of 2048 f16)
      tdm_load_2d(0, kbase + (size_t)kb * 32 * DH, 2048, 1, 2048, 1, 2048ull);
      // V^T tile: 64 rows (dims) x 32 keys, row stride NTOT (2-D DMA)
      tdm_load_2d(4096, vbase + (size_t)kb * 32, 32, 64, 2048, 64,
                  (unsigned long long)NTOT);
      __builtin_amdgcn_s_wait_tensorcnt(0);
    } else if (w == 1 && kb + 1 < NTOT / 32) {
      __builtin_prefetch(kbase + (size_t)(kb + 1) * 32 * DH, 0, 0);
    }
    __syncthreads();   // LDS tiles visible to all waves

    // K fragments from LDS: lane's column (one key), d = {0..7,16..23}, upper half +8
    F16Frag kf[2][2];
#pragma unroll
    for (int t = 0; t < 2; ++t)
#pragma unroll
      for (int dc = 0; dc < 2; ++dc) {
        const _Float16* p = ldsK + (t * 16 + lm) * DH + dc * 32 + hi * 8;
        kf[t][dc].h[0] = *(const v8h*)(p);
        kf[t][dc].h[1] = *(const v8h*)(p + 16);
      }
    v8f s0 = {}, s1 = {};
    s0 = wmma_f16(qa[0], kf[0][0], s0);
    s0 = wmma_f16(qa[1], kf[0][1], s0);
    s1 = wmma_f16(qa[0], kf[1][0], s1);
    s1 = wmma_f16(qa[1], kf[1][1], s1);

    // online softmax: rows in vgpr index, row's 16 cols in half-wave lanes
    float alpha[8];
#pragma unroll
    for (int e = 0; e < 8; ++e) {
      float sa = s0[e] * 0.125f;        // 1/sqrt(64)
      float sb = s1[e] * 0.125f;
      float m = fmaxf(sa, sb);
#pragma unroll
      for (int msk = 1; msk < 16; msk <<= 1)
        m = fmaxf(m, __shfl_xor(m, msk, 16));
      float mn = fmaxf(m_run[e], m);
      float a = __expf(m_run[e] - mn);
      float p0 = __expf(sa - mn);
      float p1 = __expf(sb - mn);
      float rs = p0 + p1;
#pragma unroll
      for (int msk = 1; msk < 16; msk <<= 1)
        rs += __shfl_xor(rs, msk, 16);
      l_run[e] = l_run[e] * a + rs;
      m_run[e] = mn;
      alpha[e] = a;
      s0[e] = p0;
      s1[e] = p1;
    }
#pragma unroll
    for (int dt = 0; dt < 4; ++dt)
#pragma unroll
      for (int e = 0; e < 8; ++e) o_acc[dt][e] *= alpha[e];

    // P: C-layout -> LDS -> A-layout (16x32 f16), wave-private staging
#pragma unroll
    for (int e = 0; e < 8; ++e) {
      int row = e + hi * 8;
      myl[row * 32 + lm] = (_Float16)s0[e];
      myl[row * 32 + 16 + lm] = (_Float16)s1[e];
    }
    F16Frag pa;
    {
      const _Float16* p = myl + lm * 32 + hi * 8;
      pa.h[0] = *(const v8h*)(p);
      pa.h[1] = *(const v8h*)(p + 16);
    }

    // O += P * V over 4 column tiles of the 64 head dims (V from LDS)
#pragma unroll
    for (int dt = 0; dt < 4; ++dt) {
      F16Frag vf;
      const _Float16* p = ldsV + (dt * 16 + lm) * 32 + hi * 8;
      vf.h[0] = *(const v8h*)(p);
      vf.h[1] = *(const v8h*)(p + 16);
      o_acc[dt] = wmma_f16(pa, vf, o_acc[dt]);
    }
    __syncthreads();   // protect LDS tiles before next iteration's DMA
  }

  float inv[8];
#pragma unroll
  for (int e = 0; e < 8; ++e) inv[e] = 1.0f / l_run[e];
#pragma unroll
  for (int dt = 0; dt < 4; ++dt)
#pragma unroll
    for (int e = 0; e < 8; ++e) {
      int ch = h * DH + dt * 16 + lm;
      int pnt = rb * 16 + e + hi * 8;
      size_t idx = ((size_t)b * OC + ch) * NTOT + pnt;
      out[idx] += o_acc[dt][e] * inv[e];
    }
}

// ---------------------------------------------------------------------------
extern "C" void kernel_launch(void* const* d_in, const int* in_sizes, int n_in,
                              void* d_out, int out_size, void* d_ws, size_t ws_size,
                              hipStream_t stream) {
  const float* pcd_up = (const float*)d_in[0];
  const float* psel   = (const float*)d_in[1];
  const float* pdrop  = (const float*)d_in[2];
  const int*   isel   = (const int*)d_in[3];
  const int*   idrop  = (const int*)d_in[4];
  const float* Wq     = (const float*)d_in[5];
  const float* Wk     = (const float*)d_in[6];
  const float* Wv     = (const float*)d_in[7];
  const float* Ws     = (const float*)d_in[8];
  float* out = (float*)d_out;

  char* ws = (char*)d_ws;
  _Float16* xT   = (_Float16*)(ws);                                // 2 MB
  _Float16* pT   = (_Float16*)(ws + (2u << 20));                   // 2 MB
  _Float16* Wall = (_Float16*)(ws + (4u << 20));                   // 256 KB
  _Float16* Qw   = (_Float16*)(ws + (4u << 20) + (256u << 10));    // 4 MB
  _Float16* Kw   = (_Float16*)(ws + (8u << 20) + (256u << 10));    // 4 MB
  _Float16* VTw  = (_Float16*)(ws + (12u << 20) + (256u << 10));   // 4 MB

  usa_prep_kernel<<<dim3(NTOT, BB), 128, 0, stream>>>(pcd_up, psel, pdrop,
                                                      isel, idrop, xT, pT);
  usa_wcvt_kernel<<<512, 256, 0, stream>>>(Wq, Wk, Wv, Ws, Wall);
  // 4 b * 64 o-tiles * 128 n-tiles = 32768 wave-tiles / 8 waves per block
  usa_proj_kernel<<<4096, 256, 0, stream>>>(xT, pT, Wall, Qw, Kw, VTw, out);
  // one block per (b,h,row-group): 4*4*16 = 256 blocks; 16 KB dynamic LDS
  usa_attn_kernel<<<256, 256, 16384, stream>>>(Qw, Kw, VTw, out);
}
